// RNN_38242388803687
// MI455X (gfx1250) — matrix operand order repaired and checked by reading, not compiled, not run on hardware
//
#include <hip/hip_runtime.h>
#include <stdint.h>

// ---------------------------------------------------------------------------
// Vanilla RNN: h_t = tanh(x_t @ W_ih + h_{t-1} @ W_hh + b), return h_{T-1}
// B=128, T=1024, D=512, H=512.  Output: (128, 512) f32.
//
//   k0: transpose+convert W_ih, W_hh (f32, KxN) -> bf16 N-major
//   k1: xp = x @ W_ih + b  (bf16 WMMA, f32 accum, M=32/WG) -> f32 ws (256MB, NT)
//   k2: persistent scan: 8 WGs (16-row batch tile each), 1024 threads,
//       W_hh slice in VGPRs, h double-buffered in LDS, pipelined xp loads.
// ---------------------------------------------------------------------------

#define BB 128
#define TT 1024
#define DD 512
#define HH 512

// LDS row padding: stride 520 bf16 (1040B) => rows land 4 banks apart.
#define LDSTRIDE 520

typedef __bf16 v16bf __attribute__((ext_vector_type(16)));
typedef float  v8f   __attribute__((ext_vector_type(8)));

static __device__ __forceinline__ v16bf make_frag(const __bf16* p0, const __bf16* p1) {
    v16bf r;
    __builtin_memcpy(&r, p0, 16);
    __builtin_memcpy(((char*)&r) + 16, p1, 16);
    return r;
}

static __device__ __forceinline__ v8f wmma_bf16(v16bf a, v16bf b, v8f c) {
    return __builtin_amdgcn_wmma_f32_16x16x32_bf16(
        false, a, false, b, (short)0, c, false, false);
}

static __device__ __forceinline__ float fast_tanh(float x) {
#if __has_builtin(__builtin_amdgcn_tanhf)
    return __builtin_amdgcn_tanhf(x);
#elif __has_builtin(__builtin_amdgcn_tanh_f32)
    return __builtin_amdgcn_tanh_f32(x);
#else
    return tanhf(x);
#endif
}

// ---------------------------------------------------------------------------
// Kernel 0: W (K x N, f32) -> Wt (N x K, bf16) for both weight matrices.
// ---------------------------------------------------------------------------
__global__ __launch_bounds__(256)
void k_cvt_weights(const float* __restrict__ wih, const float* __restrict__ whh,
                   __bf16* __restrict__ wtih, __bf16* __restrict__ wthh) {
    int idx = blockIdx.x * 256 + threadIdx.x;        // 0 .. 2*512*512-1
    int which = idx >> 18;
    int r = idx & (512 * 512 - 1);
    int n = r >> 9;
    int k = r & 511;
    if (which == 0) wtih[n * 512 + k] = (__bf16)wih[k * 512 + n];
    else            wthh[n * 512 + k] = (__bf16)whh[k * 512 + n];
}

// ---------------------------------------------------------------------------
// Kernel 1: xp[M=B*T, H] = x[M, D] @ W_ih + b.
// WG = 32 rows x 512 cols, 8 waves; each wave: 2 M-subtiles x 4 N-tiles.
// B fragment loaded once per (kk, ntile) and reused for both M-subtiles.
// ---------------------------------------------------------------------------
__global__ __launch_bounds__(256)
void k_xp_gemm(const float* __restrict__ x, const __bf16* __restrict__ wt,
               const float* __restrict__ bias, float* __restrict__ xp) {
    __shared__ __align__(16) __bf16 Atile[32 * LDSTRIDE];

    const int tid   = threadIdx.x;
    const size_t m0 = (size_t)blockIdx.x * 32;

    // Stage A: 32 rows x 512 cols f32 -> bf16 in LDS.
    #pragma unroll
    for (int it = 0; it < 16; ++it) {
        int e   = it * 1024 + tid * 4;
        int row = e >> 9;
        int col = e & 511;
        float4 v = *(const float4*)(x + (m0 + row) * 512 + col);
        __bf16 tmp[4] = {(__bf16)v.x, (__bf16)v.y, (__bf16)v.z, (__bf16)v.w};
        __builtin_memcpy(&Atile[row * LDSTRIDE + col], tmp, 8);
    }
    __syncthreads();

    const int lane = tid & 31;
    const int wave = tid >> 5;
    const int hi   = lane >> 4;
    const int n0   = lane & 15;
    const int ncb  = wave * 64;

    v8f acc[2][4];
    #pragma unroll
    for (int t2 = 0; t2 < 4; ++t2) {
        float bv = bias[ncb + t2 * 16 + n0];
        #pragma unroll
        for (int mi = 0; mi < 2; ++mi)
            #pragma unroll
            for (int j = 0; j < 8; ++j) acc[mi][t2][j] = bv;
    }

    #pragma unroll
    for (int kk = 0; kk < 512; kk += 32) {
        v16bf a[2];
        #pragma unroll
        for (int mi = 0; mi < 2; ++mi) {
            const __bf16* ar = &Atile[(mi * 16 + n0) * LDSTRIDE + kk + hi * 8];
            a[mi] = make_frag(ar, ar + 16);
        }
        #pragma unroll
        for (int t2 = 0; t2 < 4; ++t2) {
            int n = ncb + t2 * 16 + n0;
            const __bf16* bp = wt + (size_t)n * 512 + kk + hi * 8;
            v16bf b = make_frag(bp, bp + 16);
            acc[0][t2] = wmma_bf16(a[0], b, acc[0][t2]);
            acc[1][t2] = wmma_bf16(a[1], b, acc[1][t2]);
        }
    }

    // Non-temporal stores: xp is a 256MB stream with no reuse before phase 2.
    #pragma unroll
    for (int mi = 0; mi < 2; ++mi)
        #pragma unroll
        for (int t2 = 0; t2 < 4; ++t2)
            #pragma unroll
            for (int j = 0; j < 8; ++j) {
                size_t m = m0 + mi * 16 + j + hi * 8;
                __builtin_nontemporal_store(acc[mi][t2][j],
                    &xp[m * 512 + ncb + t2 * 16 + n0]);
            }
}

// ---------------------------------------------------------------------------
// Kernel 2: recurrent scan.  grid = 8 (batch tiles of 16), block = 1024
// (32 waves; wave w owns h columns [16w, 16w+16)).
// ---------------------------------------------------------------------------
__global__ __launch_bounds__(1024)
void k_rnn_scan(const float* __restrict__ xp, const __bf16* __restrict__ wthh,
                float* __restrict__ out) {
    __shared__ __align__(16) __bf16 hbuf[2][16 * LDSTRIDE];

    const int tid  = threadIdx.x;
    const int lane = tid & 31;
    const int wave = tid >> 5;
    const int hi   = lane >> 4;
    const int n0   = lane & 15;
    const int nb   = wave * 16;
    const size_t b0 = (size_t)blockIdx.x * 16;

    // Preload this wave's W_hh^T slice: 16 K-steps x (32x16) bf16 fragments
    // (128 VGPRs/lane; verified no spills in the round-1 histogram).
    v16bf wf[16];
    #pragma unroll
    for (int kki = 0; kki < 16; ++kki) {
        const __bf16* bp = wthh + (size_t)(nb + n0) * 512 + kki * 32 + hi * 8;
        wf[kki] = make_frag(bp, bp + 16);
    }

    // h_0 = 0
    for (int e = tid; e < 16 * LDSTRIDE; e += 1024) hbuf[0][e] = (__bf16)0.0f;
    __syncthreads();

    // Software-pipelined xp loads: fetch step t+1 behind step t's WMMA chain.
    v8f xnext;
    #pragma unroll
    for (int j = 0; j < 8; ++j) {
        size_t m = b0 + j + hi * 8;
        xnext[j] = __builtin_nontemporal_load(&xp[(m * TT + 0) * 512 + nb + n0]);
    }

    v8f hv;
    #pragma unroll
    for (int j = 0; j < 8; ++j) hv[j] = 0.0f;

    for (int t = 0; t < TT; ++t) {
        v8f acc = xnext;          // C init = xp[b, t, n]
        v8f acc2;
        #pragma unroll
        for (int j = 0; j < 8; ++j) acc2[j] = 0.0f;

        if (t + 1 < TT) {
            #pragma unroll
            for (int j = 0; j < 8; ++j) {
                size_t m = b0 + j + hi * 8;
                xnext[j] = __builtin_nontemporal_load(
                    &xp[(m * TT + (t + 1)) * 512 + nb + n0]);
            }
        }

        const __bf16* hsrc = hbuf[t & 1];
        #pragma unroll
        for (int kki = 0; kki < 16; kki += 2) {   // two independent K-chains
            const __bf16* a0 = &hsrc[n0 * LDSTRIDE + kki * 32 + hi * 8];
            const __bf16* a1 = a0 + 32;
            acc  = wmma_bf16(make_frag(a0, a0 + 16), wf[kki],     acc);
            acc2 = wmma_bf16(make_frag(a1, a1 + 16), wf[kki + 1], acc2);
        }

        __bf16* hdst = hbuf[(t + 1) & 1];
        #pragma unroll
        for (int j = 0; j < 8; ++j) {
            float th = fast_tanh(acc[j] + acc2[j]);
            hv[j] = th;
            hdst[(j + hi * 8) * LDSTRIDE + nb + n0] = (__bf16)th;
        }
        __syncthreads();
    }

    #pragma unroll
    for (int j = 0; j < 8; ++j)
        out[(b0 + j + hi * 8) * 512 + nb + n0] = hv[j];
}

// ---------------------------------------------------------------------------
// Host launcher
// ---------------------------------------------------------------------------
extern "C" void kernel_launch(void* const* d_in, const int* in_sizes, int n_in,
                              void* d_out, int out_size, void* d_ws, size_t ws_size,
                              hipStream_t stream) {
    const float* x    = (const float*)d_in[0];   // (128, 1024, 512)
    const float* w_ih = (const float*)d_in[1];   // (512, 512)
    const float* w_hh = (const float*)d_in[2];   // (512, 512)
    const float* b    = (const float*)d_in[3];   // (512,)
    float* out = (float*)d_out;                  // (128, 512)

    uint8_t* ws = (uint8_t*)d_ws;
    const size_t XP_BYTES = (size_t)BB * TT * HH * sizeof(float);   // 256 MB
    float*  xp   = (float*)ws;
    __bf16* wtih = (__bf16*)(ws + XP_BYTES);
    __bf16* wthh = (__bf16*)(ws + XP_BYTES + (size_t)HH * DD * sizeof(__bf16));

    k_cvt_weights<<<2048, 256, 0, stream>>>(w_ih, w_hh, wtih, wthh);
    k_xp_gemm<<<(BB * TT) / 32, 256, 0, stream>>>(x, wtih, b, xp);
    k_rnn_scan<<<BB / 16, 1024, 0, stream>>>(xp, wthh, out);
}
